// MultiQueryAttention_82386062672302
// MI455X (gfx1250) — matrix operand order repaired and checked by reading, not compiled, not run on hardware
//
#include <hip/hip_runtime.h>
#include <math.h>

// ---------------------------------------------------------------------------
// MQA: B=2, S=2048, D_MODEL=2048, N_HEAD=16, D_K=128, causal, RoPE, MQA (1 KV head)
// bf16 WMMA (v_wmma_f32_16x16x32_bf16) for all matmuls; flash-style online
// softmax; GEMM B-tiles staged in LDS (async-to-LDS when available) with
// double buffering + software-pipelined A-fragment loads.
// ---------------------------------------------------------------------------

#define B_ 2
#define S_ 2048
#define DM_ 2048
#define NH_ 16
#define DK_ 128

typedef __attribute__((ext_vector_type(16))) __bf16 bf16x16;
typedef __attribute__((ext_vector_type(8)))  float  f32x8;

union FragU { bf16x16 v; uint4 u[2]; };

#if defined(__has_builtin)
#if __has_builtin(__builtin_amdgcn_global_load_async_to_lds_b128)
#define USE_ASYNC_LDS 1
#else
#define USE_ASYNC_LDS 0
#endif
#else
#define USE_ASYNC_LDS 0
#endif

#if USE_ASYNC_LDS
typedef int i32x4_gcc __attribute__((vector_size(16)));
typedef __attribute__((address_space(1))) i32x4_gcc glob_i32x4;
typedef __attribute__((address_space(3))) i32x4_gcc lds_i32x4;
#endif

__device__ __forceinline__ unsigned short f32_to_bf16(float f) {
  unsigned int u = __float_as_uint(f);
  u += 0x7FFFu + ((u >> 16) & 1u);   // round to nearest even
  return (unsigned short)(u >> 16);
}

// Copy 16 bytes global -> LDS. Async (ASYNCcnt-tracked) when the builtin
// exists; otherwise plain load + ds_store.
__device__ __forceinline__ void stage_b128(const void* gsrc, void* ldst) {
#if USE_ASYNC_LDS
  __builtin_amdgcn_global_load_async_to_lds_b128(
      (glob_i32x4*)(gsrc), (lds_i32x4*)(ldst), 0, 0);
#else
  *(uint4*)ldst = *(const uint4*)gsrc;
#endif
}

// A-fragment (16x32, MxK): lane<16 holds row M=lane, K={k0..k0+7, k0+16..k0+23};
// lane>=16 holds row M=lane-16, K={k0+8..k0+15, k0+24..k0+31}.
__device__ __forceinline__ bf16x16 load_a_frag(const unsigned short* base, int ld,
                                               int row0, int k0, int lane) {
  int r = lane & 15, sel = (lane >> 4) & 1;
  const unsigned short* p = base + (size_t)(row0 + r) * ld + k0 + sel * 8;
  FragU f;
  f.u[0] = *(const uint4*)(p);
  f.u[1] = *(const uint4*)(p + 16);
  return f.v;
}

// B-fragment (32x16, KxN): lane<16 holds col N=lane, K={k0..k0+7, k0+16..k0+23};
// lane>=16 holds col N=lane-16, K={k0+8..k0+15, k0+24..k0+31}. Bt is N-major (N x K, ld=K).
__device__ __forceinline__ bf16x16 load_b_frag(const unsigned short* base, int ld,
                                               int n0, int k0, int lane) {
  int n = lane & 15, sel = (lane >> 4) & 1;
  const unsigned short* p = base + (size_t)(n0 + n) * ld + k0 + sel * 8;
  FragU f;
  f.u[0] = *(const uint4*)(p);
  f.u[1] = *(const uint4*)(p + 16);
  return f.v;
}

// ------------------------- elementwise / layout kernels ---------------------

__global__ __launch_bounds__(256) void cvt_bf16_kernel(const float* x,
                                                       unsigned short* y, int n) {
  int i = blockIdx.x * 256 + threadIdx.x;
  if (i < n) y[i] = f32_to_bf16(x[i]);
}

// W (K x N) row-major f32  ->  Wt (N x K) row-major bf16
__global__ __launch_bounds__(256) void wtrans_kernel(const float* W,
                                                     unsigned short* Wt,
                                                     int K, int N) {
  int i = blockIdx.x * 256 + threadIdx.x;
  if (i >= K * N) return;
  int n = i % N, k = i / N;
  Wt[(size_t)n * K + k] = f32_to_bf16(W[i]);
}

// RoPE on q (B,S,2048 f32) per head of 128, pre-scaled by 1/sqrt(128) -> bf16
__global__ __launch_bounds__(256) void rope_q_kernel(const float* q, const int* pos,
                                                     unsigned short* qb) {
  int tid = blockIdx.x * 256 + threadIdx.x;          // B*S*NH*64 threads
  int i = tid & 63;
  int h = (tid >> 6) & 15;
  int s = (tid >> 10) & 2047;
  int b = tid >> 21;
  size_t base = ((size_t)(b * S_ + s)) * DM_ + h * DK_ + 2 * i;
  float inv_freq = __expf(-(float)(2 * i) * (9.210340371976184f / 128.0f)); // ln(1e4)
  float ang = (float)pos[s] * inv_freq;
  float c = cosf(ang), sn = sinf(ang);
  float x1 = q[base], x2 = q[base + 1];
  const float sc = 0.08838834764831845f;             // 1/sqrt(128)
  qb[base]     = f32_to_bf16((x1 * c - x2 * sn) * sc);
  qb[base + 1] = f32_to_bf16((x1 * sn + x2 * c) * sc);
}

// RoPE on k (B,S,128 f32) -> bf16
__global__ __launch_bounds__(256) void rope_k_kernel(const float* k, const int* pos,
                                                     unsigned short* kb) {
  int tid = blockIdx.x * 256 + threadIdx.x;          // B*S*64 threads
  int i = tid & 63;
  int s = (tid >> 6) & 2047;
  int b = tid >> 17;
  size_t base = ((size_t)(b * S_ + s)) * DK_ + 2 * i;
  float inv_freq = __expf(-(float)(2 * i) * (9.210340371976184f / 128.0f));
  float ang = (float)pos[s] * inv_freq;
  float c = cosf(ang), sn = sinf(ang);
  float x1 = k[base], x2 = k[base + 1];
  kb[base]     = f32_to_bf16(x1 * c - x2 * sn);
  kb[base + 1] = f32_to_bf16(x1 * sn + x2 * c);
}

// v (B,S,128 f32) -> vt (B,128,S bf16)
__global__ __launch_bounds__(256) void vtr_kernel(const float* v, unsigned short* vt) {
  int tid = blockIdx.x * 256 + threadIdx.x;          // B*S*128 threads
  int d = tid & 127;
  int s = (tid >> 7) & 2047;
  int b = tid >> 18;
  vt[((size_t)b * DK_ + d) * S_ + s] = f32_to_bf16(v[tid]);
}

// ------------------------------ WMMA GEMM ----------------------------------
// C(MxN) = A(MxK,bf16) @ Bt(NxK,bf16)^T + bias.
// Block: 8 waves = 128(M) x 64(N); K step 32. B tile (64x32) is shared by all
// waves -> staged once per block into LDS (double buffered, async when
// available); per-wave A fragment is software-pipelined one K-step ahead.
__global__ __launch_bounds__(256) void gemm_bf16_kernel(
    const unsigned short* __restrict__ A, const unsigned short* __restrict__ Bt,
    const float* __restrict__ bias, float* Cf, unsigned short* Cb,
    int M, int N, int K) {
  __shared__ unsigned short Bs[2][64 * 32];

  int tid = threadIdx.x;
  int lane = tid & 31;
  int wave = tid >> 5;
  int m0 = blockIdx.y * 128 + wave * 16;
  int n0 = blockIdx.x * 64;

  // this thread's 16-byte slice of the 64x32 staging tile
  int sn = tid >> 2;            // 0..63 (row n within tile)
  int sk = (tid & 3) * 8;       // 0,8,16,24 (k within tile)
  const unsigned short* sBase = Bt + (size_t)(n0 + sn) * K + sk;

  f32x8 acc[4];
#pragma unroll
  for (int t = 0; t < 4; ++t) acc[t] = f32x8{0.f,0.f,0.f,0.f,0.f,0.f,0.f,0.f};

  int nK = K >> 5;
  stage_b128(sBase, &Bs[0][sn * 32 + sk]);
  bf16x16 a = load_a_frag(A, K, m0, 0, lane);

  for (int it = 0; it < nK; ++it) {
    int cur = it & 1;
    bf16x16 a_cur = a;
    if (it + 1 < nK) {
      stage_b128(sBase + (size_t)(it + 1) * 32, &Bs[cur ^ 1][sn * 32 + sk]);
      a = load_a_frag(A, K, m0, (it + 1) * 32, lane);   // prefetch next A
      asm volatile("s_wait_asynccnt 0x1" ::: "memory"); // oldest stage done
    } else {
      asm volatile("s_wait_asynccnt 0x0" ::: "memory");
    }
    __syncthreads();                                    // publish Bs[cur]
#pragma unroll
    for (int t = 0; t < 4; ++t) {
      bf16x16 b = load_b_frag(&Bs[cur][0], 32, t * 16, 0, lane);
      acc[t] = __builtin_amdgcn_wmma_f32_16x16x32_bf16(
          false, a_cur, false, b, (short)0, acc[t], false, false);
    }
    __syncthreads();                                    // Bs[cur] reusable
  }

  int cn = lane & 15, rb = (lane >> 4) * 8;
#pragma unroll
  for (int t = 0; t < 4; ++t) {
    int col = n0 + t * 16 + cn;
    float bv = bias ? bias[col] : 0.f;
#pragma unroll
    for (int r = 0; r < 8; ++r) {
      size_t idx = (size_t)(m0 + rb + r) * N + col;
      float val = acc[t][r] + bv;
      if (Cf) Cf[idx] = val;
      if (Cb) Cb[idx] = f32_to_bf16(val);
    }
  }
}

// --------------------------- flash MQA kernel ------------------------------
__global__ __launch_bounds__(256) void mqa_flash_kernel(
    const unsigned short* __restrict__ Q,   // (B,S,2048) roped, pre-scaled
    const unsigned short* __restrict__ Kr,  // (B,S,128)  roped
    const unsigned short* __restrict__ Vt,  // (B,128,S)
    unsigned short* __restrict__ Attn) {    // (B,S,2048) bf16
  __shared__ unsigned short lds_p[8][16 * 32];

  int lane = threadIdx.x & 31;
  int wave = threadIdx.x >> 5;
  int m0 = blockIdx.x * 128 + wave * 16;
  int h = blockIdx.y;
  int b = blockIdx.z;

  const unsigned short* Qb = Q + ((size_t)b * S_) * DM_ + h * DK_;
  const unsigned short* Kb = Kr + (size_t)b * S_ * DK_;
  const unsigned short* Vb = Vt + (size_t)b * DK_ * S_;

  bf16x16 qf[4];
#pragma unroll
  for (int c = 0; c < 4; ++c) qf[c] = load_a_frag(Qb, DM_, m0, c * 32, lane);

  f32x8 acc[8];
#pragma unroll
  for (int t = 0; t < 8; ++t) acc[t] = f32x8{0.f,0.f,0.f,0.f,0.f,0.f,0.f,0.f};
  float m_run[8], l_run[8];
#pragma unroll
  for (int r = 0; r < 8; ++r) { m_run[r] = -__builtin_inff(); l_run[r] = 0.f; }

  int cn = lane & 15, rb = (lane >> 4) * 8;
  int nIter = (m0 + 16 + 31) >> 5;                    // causal: keys 0..m0+15
  unsigned short* myp = &lds_p[wave][0];

  for (int j = 0; j < nIter; ++j) {
    int kk0 = j << 5;
    f32x8 s0 = f32x8{0.f,0.f,0.f,0.f,0.f,0.f,0.f,0.f};
    f32x8 s1 = f32x8{0.f,0.f,0.f,0.f,0.f,0.f,0.f,0.f};
#pragma unroll
    for (int c = 0; c < 4; ++c) {
      bf16x16 kb0 = load_b_frag(Kb, DK_, kk0,      c * 32, lane);
      bf16x16 kb1 = load_b_frag(Kb, DK_, kk0 + 16, c * 32, lane);
      s0 = __builtin_amdgcn_wmma_f32_16x16x32_bf16(false, qf[c], false, kb0,
                                                   (short)0, s0, false, false);
      s1 = __builtin_amdgcn_wmma_f32_16x16x32_bf16(false, qf[c], false, kb1,
                                                   (short)0, s1, false, false);
    }

    // causal mask + row max (rows live in 16-lane halves, one row per C-VGPR)
    float mx[8];
#pragma unroll
    for (int r = 0; r < 8; ++r) {
      int rowg = m0 + rb + r;
      float a0 = (kk0 + cn      <= rowg) ? s0[r] : -__builtin_inff();
      float a1 = (kk0 + 16 + cn <= rowg) ? s1[r] : -__builtin_inff();
      s0[r] = a0; s1[r] = a1;
      mx[r] = fmaxf(a0, a1);
    }
#pragma unroll
    for (int off = 1; off < 16; off <<= 1) {
#pragma unroll
      for (int r = 0; r < 8; ++r) mx[r] = fmaxf(mx[r], __shfl_xor(mx[r], off));
    }

    float al[8], rs[8];
#pragma unroll
    for (int r = 0; r < 8; ++r) {
      float mnew = fmaxf(m_run[r], mx[r]);
      al[r] = __expf(m_run[r] - mnew);                // exp(-inf)=0 on 1st iter
      float p0 = __expf(s0[r] - mnew);
      float p1 = __expf(s1[r] - mnew);
      m_run[r] = mnew;
      rs[r] = p0 + p1;
      myp[(rb + r) * 32 + cn]      = f32_to_bf16(p0);
      myp[(rb + r) * 32 + 16 + cn] = f32_to_bf16(p1);
    }
#pragma unroll
    for (int off = 1; off < 16; off <<= 1) {
#pragma unroll
      for (int r = 0; r < 8; ++r) rs[r] += __shfl_xor(rs[r], off);
    }
#pragma unroll
    for (int r = 0; r < 8; ++r) l_run[r] = l_run[r] * al[r] + rs[r];
#pragma unroll
    for (int t = 0; t < 8; ++t)
#pragma unroll
      for (int r = 0; r < 8; ++r) acc[t][r] *= al[r];

    // wave-private LDS round trip: C-layout P -> A-fragment P (16x32 bf16)
    asm volatile("s_wait_dscnt 0x0" ::: "memory");
    bf16x16 pf = load_a_frag(myp, 32, 0, 0, lane);

#pragma unroll
    for (int t = 0; t < 8; ++t) {
      bf16x16 vb = load_b_frag(Vb, S_, t * 16, kk0, lane);  // Vt rows = d
      acc[t] = __builtin_amdgcn_wmma_f32_16x16x32_bf16(
          false, pf, false, vb, (short)0, acc[t], false, false);
    }
  }

#pragma unroll
  for (int r = 0; r < 8; ++r) l_run[r] = 1.0f / l_run[r];
  unsigned short* Ab = Attn + ((size_t)b * S_) * DM_ + h * DK_;
#pragma unroll
  for (int t = 0; t < 8; ++t)
#pragma unroll
    for (int r = 0; r < 8; ++r)
      Ab[(size_t)(m0 + rb + r) * DM_ + t * 16 + cn] =
          f32_to_bf16(acc[t][r] * l_run[r]);
}

// ------------------------------- launch ------------------------------------

extern "C" void kernel_launch(void* const* d_in, const int* in_sizes, int n_in,
                              void* d_out, int out_size, void* d_ws, size_t ws_size,
                              hipStream_t stream) {
  const float* x   = (const float*)d_in[0];
  const int*   pos = (const int*)d_in[1];
  const float* Wq  = (const float*)d_in[2];
  const float* bq  = (const float*)d_in[3];
  const float* Wk  = (const float*)d_in[4];
  const float* bk  = (const float*)d_in[5];
  const float* Wv  = (const float*)d_in[6];
  const float* bv  = (const float*)d_in[7];
  const float* Wo  = (const float*)d_in[8];
  const float* bo  = (const float*)d_in[9];
  float* out = (float*)d_out;

  unsigned char* w = (unsigned char*)d_ws;
  size_t off = 0;
  auto alloc = [&](size_t bytes) {
    size_t o = off; off += (bytes + 255) & ~(size_t)255; return o;
  };
  unsigned short* x_bf  = (unsigned short*)(w + alloc((size_t)B_*S_*DM_*2));
  unsigned short* wq_t  = (unsigned short*)(w + alloc((size_t)DM_*DM_*2));
  unsigned short* wk_t  = (unsigned short*)(w + alloc((size_t)DM_*DK_*2));
  unsigned short* wv_t  = (unsigned short*)(w + alloc((size_t)DM_*DK_*2));
  unsigned short* wo_t  = (unsigned short*)(w + alloc((size_t)DM_*DM_*2));
  float*          q_f32 = (float*)(w + alloc((size_t)B_*S_*DM_*4));
  float*          k_f32 = (float*)(w + alloc((size_t)B_*S_*DK_*4));
  float*          v_f32 = (float*)(w + alloc((size_t)B_*S_*DK_*4));
  unsigned short* q_bf  = (unsigned short*)(w + alloc((size_t)B_*S_*DM_*2));
  unsigned short* k_bf  = (unsigned short*)(w + alloc((size_t)B_*S_*DK_*2));
  unsigned short* vt_bf = (unsigned short*)(w + alloc((size_t)B_*S_*DK_*2));
  unsigned short* at_bf = (unsigned short*)q_f32;  // alias: q_f32 dead by then

  const int M = B_ * S_;  // 4096

  // 1) operand conversion / weight transposes
  {
    int n = B_ * S_ * DM_;
    cvt_bf16_kernel<<<(n + 255) / 256, 256, 0, stream>>>(x, x_bf, n);
    wtrans_kernel<<<(DM_*DM_ + 255) / 256, 256, 0, stream>>>(Wq, wq_t, DM_, DM_);
    wtrans_kernel<<<(DM_*DK_ + 255) / 256, 256, 0, stream>>>(Wk, wk_t, DM_, DK_);
    wtrans_kernel<<<(DM_*DK_ + 255) / 256, 256, 0, stream>>>(Wv, wv_t, DM_, DK_);
    wtrans_kernel<<<(DM_*DM_ + 255) / 256, 256, 0, stream>>>(Wo, wo_t, DM_, DM_);
  }

  // 2) projections: q = x@Wq+bq ; k = x@Wk+bk ; v = x@Wv+bv
  gemm_bf16_kernel<<<dim3(DM_/64, M/128), 256, 0, stream>>>(
      x_bf, wq_t, bq, q_f32, nullptr, M, DM_, DM_);
  gemm_bf16_kernel<<<dim3(DK_/64, M/128), 256, 0, stream>>>(
      x_bf, wk_t, bk, k_f32, nullptr, M, DK_, DM_);
  gemm_bf16_kernel<<<dim3(DK_/64, M/128), 256, 0, stream>>>(
      x_bf, wv_t, bv, v_f32, nullptr, M, DK_, DM_);

  // 3) RoPE + layout conversion to bf16
  rope_q_kernel<<<(B_*S_*NH_*64) / 256, 256, 0, stream>>>(q_f32, pos, q_bf);
  rope_k_kernel<<<(B_*S_*64) / 256, 256, 0, stream>>>(k_f32, pos, k_bf);
  vtr_kernel<<<(B_*S_*DK_) / 256, 256, 0, stream>>>(v_f32, vt_bf);

  // 4) flash attention (causal, MQA) -> attn bf16 (B,S,2048)
  mqa_flash_kernel<<<dim3(S_/128, NH_, B_), 256, 0, stream>>>(
      q_bf, k_bf, vt_bf, at_bf);

  // 5) output projection: out = attn @ Wo + bo (f32)
  gemm_bf16_kernel<<<dim3(DM_/64, M/128), 256, 0, stream>>>(
      at_bf, wo_t, bo, out, nullptr, M, DM_, DM_);
}
